// CoreCoPER_71253507441311
// MI455X (gfx1250) — matrix operand order repaired and checked by reading, not compiled, not run on hardware
//
#include <hip/hip_runtime.h>
#include <hip/hip_bf16.h>

// ---------------------------------------------------------------------------
// Problem constants (from reference)
// ---------------------------------------------------------------------------
#define BATCH    256
#define FEAT_IN  200
#define FEAT_OUT 200
#define GEN_DIM  200
#define HID      200
#define HIDP     224          // HID padded to 7 * 32 for bf16 WMMA K-steps
#define KSTEPS   7            // 224 / 32
#define NC       32
#define OH       8
#define OW       18
#define FC_IN    4608         // NC * OH * OW
#define ICHUNK   128
#define ICHUNKS  36           // 4608 / 128
#define OTILES   13           // ceil(200 / 16)
#define W2COLS   921600       // FC_IN * FEAT_OUT
#define KPERT    14           // 224 / 16 staged elements per thread

typedef __attribute__((ext_vector_type(16))) __bf16 bf16x16;
typedef __attribute__((ext_vector_type(8)))  float  floatx8;

union ABFrag {
    bf16x16 v;
    uint4   q[2];
};

static __device__ __forceinline__ unsigned short f2bf(float f) {
    union { float f; unsigned u; } x; x.f = f;
    unsigned u = x.u;
    u += 0x7FFFu + ((u >> 16) & 1u);   // round-to-nearest-even
    return (unsigned short)(u >> 16);
}

// ---------------------------------------------------------------------------
// Small generators: out = relu(relu(r@W1+b1) @ W2 + b2). One block per sample.
// ---------------------------------------------------------------------------
__global__ void gen_small_kernel(const float* __restrict__ r,
                                 const float* __restrict__ W1,
                                 const float* __restrict__ b1,
                                 const float* __restrict__ W2,
                                 const float* __restrict__ b2,
                                 float* __restrict__ out, int od) {
    const int b = blockIdx.x;
    const int t = threadIdx.x;
    __shared__ float rs[GEN_DIM];
    __shared__ float hs[HID];
    if (t < GEN_DIM) rs[t] = r[b * GEN_DIM + t];
    __syncthreads();
    if (t < HID) {
        float s = b1[t];
        #pragma unroll 4
        for (int k = 0; k < GEN_DIM; ++k) s = fmaf(rs[k], W1[k * HID + t], s);
        hs[t] = s > 0.f ? s : 0.f;
    }
    __syncthreads();
    for (int c = t; c < od; c += blockDim.x) {
        float s = b2[c];
        #pragma unroll 4
        for (int k = 0; k < HID; ++k) s = fmaf(hs[k], W2[(size_t)k * od + c], s);
        out[(size_t)b * od + c] = s > 0.f ? s : 0.f;
    }
}

// ---------------------------------------------------------------------------
// fw hidden layer: H[b][k] = relu(r_b @ fw_W1[:,k] + b1[k]) as bf16, K padded
// ---------------------------------------------------------------------------
__global__ void fw_hidden_kernel(const float* __restrict__ r,
                                 const float* __restrict__ W1,
                                 const float* __restrict__ b1,
                                 unsigned short* __restrict__ H) {
    const int b = blockIdx.x;
    const int t = threadIdx.x;
    __shared__ float rs[GEN_DIM];
    if (t < GEN_DIM) rs[t] = r[b * GEN_DIM + t];
    __syncthreads();
    if (t < HIDP) {
        float v = 0.f;
        if (t < HID) {
            float s = b1[t];
            #pragma unroll 4
            for (int k = 0; k < GEN_DIM; ++k) s = fmaf(rs[k], W1[k * HID + t], s);
            v = s > 0.f ? s : 0.f;
        }
        H[b * HIDP + t] = f2bf(v);
    }
}

// ---------------------------------------------------------------------------
// Per-sample grouped conv + relu -> xT[i][b]  (transposed for coalesced reads)
// ---------------------------------------------------------------------------
__global__ void conv_kernel(const float* __restrict__ img_all,
                            const float* __restrict__ cw,   // (B, 288)
                            const float* __restrict__ cb,   // (B, 32)
                            float* __restrict__ xT) {       // (FC_IN, B)
    const int b = blockIdx.x;
    const int t = threadIdx.x;
    __shared__ float img[FEAT_IN];
    __shared__ float w[NC * 9];
    __shared__ float bias[NC];
    if (t < FEAT_IN) img[t] = img_all[b * FEAT_IN + t];
    for (int j = t; j < NC * 9; j += blockDim.x) w[j] = cw[(size_t)b * NC * 9 + j];
    if (t < NC) bias[t] = cb[b * NC + t];
    __syncthreads();
    for (int i = t; i < FC_IN; i += blockDim.x) {
        const int c  = i / (OH * OW);
        const int rm = i % (OH * OW);
        const int oh = rm / OW;
        const int ow = rm % OW;
        float s = bias[c];
        #pragma unroll
        for (int kh = 0; kh < 3; ++kh)
            #pragma unroll
            for (int kw = 0; kw < 3; ++kw)
                s = fmaf(img[(oh + kh) * 20 + (ow + kw)], w[c * 9 + kh * 3 + kw], s);
        s = s > 0.f ? s : 0.f;
        xT[(size_t)i * BATCH + b] = s;
    }
}

__global__ void zero_kernel(float* __restrict__ p, int n) {
    int i = blockIdx.x * blockDim.x + threadIdx.x;
    if (i < n) p[i] = 0.f;
}

// ---------------------------------------------------------------------------
// Fused WMMA kernel (double-buffered, fully register-pipelined scalars):
//   acc[b,o] += sum_i x[b,i] * relu( (H @ W2)[b, i*200+o] + b2[i*200+o] )
// grid = (OTILES, ICHUNKS), block = 256 threads = 8 waves.
// Each wave owns M-tiles {wave, wave+8}. A-fragments (H) are i-invariant and
// hoisted into VGPRs. Per i: one barrier; global loads (W2 slab, x column,
// b2 scalar) for i+1 issue before the 14 WMMAs of i; the epilogue consumes
// only values fetched one iteration earlier, so its load-wait never blocks
// on the in-flight prefetch; convert+LDS-store fills the other buffer after
// the compute.
// ---------------------------------------------------------------------------
__global__ void __launch_bounds__(256)
fused_einsum_kernel(const unsigned short* __restrict__ Hbf, // (B, HIDP) bf16
                    const float* __restrict__ fwW2,         // (HID, W2COLS)
                    const float* __restrict__ fwb2,         // (W2COLS)
                    const float* __restrict__ xT,           // (FC_IN, B)
                    float* __restrict__ accg) {             // (B, FEAT_OUT)
    __shared__ __align__(16) unsigned short Bs[2][16 * HIDP]; // [buf][o][k] bf16
    __shared__ float xcolS[2][BATCH];

    const int tid  = threadIdx.x;
    const int lane = tid & 31;
    const int wave = tid >> 5;
    const int oL   = lane & 15;
    const int Kh   = lane >> 4;       // 0 or 1
    const int o0   = blockIdx.x * 16;
    const int i0   = blockIdx.y * ICHUNK;
    const int iend = i0 + ICHUNK;

    // staging role of this thread
    const int  so   = tid & 15;       // o within tile
    const int  skb  = tid >> 4;       // k base 0..15
    const bool ok_o  = (o0 + so) < FEAT_OUT;
    const bool ok_oL = (o0 + oL) < FEAT_OUT;

    // ---- load A fragments (H) once: 2 M-tiles x 7 K-steps ------------------
    ABFrag afrag[2][KSTEPS];
    #pragma unroll
    for (int mt = 0; mt < 2; ++mt) {
        const int b = (wave + mt * 8) * 16 + oL;
        const uint4* hrow = (const uint4*)(Hbf + (size_t)b * HIDP);
        #pragma unroll
        for (int kk = 0; kk < KSTEPS; ++kk) {
            afrag[mt][kk].q[0] = hrow[kk * 4 + Kh];       // K = 32kk + 8*Kh + 0..7
            afrag[mt][kk].q[1] = hrow[kk * 4 + 2 + Kh];   // K = 32kk + 16 + 8*Kh + 0..7
        }
    }

    float acc[2][8];
    #pragma unroll
    for (int mt = 0; mt < 2; ++mt)
        #pragma unroll
        for (int j = 0; j < 8; ++j) acc[mt][j] = 0.f;

    // ---- prologue: stage buffer 0 for i0; b2 for i0 into register ----------
    float vr[KPERT];
    float xv;
    float b2cur;
    {
        const float* col = fwW2 + (size_t)i0 * FEAT_OUT + (o0 + so);
        #pragma unroll
        for (int j = 0; j < KPERT; ++j) {
            const int k = skb + j * 16;
            vr[j] = (ok_o && k < HID) ? col[(size_t)k * W2COLS] : 0.f;
        }
        xv    = xT[(size_t)i0 * BATCH + tid];
        b2cur = ok_oL ? fwb2[(size_t)i0 * FEAT_OUT + o0 + oL] : 0.f;
        #pragma unroll
        for (int j = 0; j < KPERT; ++j)
            Bs[0][so * HIDP + (skb + j * 16)] = f2bf(vr[j]);
        xcolS[0][tid] = xv;
    }

    for (int i = i0; i < iend; ++i) {
        const int  cur      = (i - i0) & 1;
        const bool havenext = (i + 1) < iend;
        __syncthreads();   // buffer `cur` published; buffer `cur^1` free to fill

        // ---- issue global loads for i+1 (latency hidden behind WMMAs) ------
        float b2n = 0.f;
        if (havenext) {
            const float* col = fwW2 + (size_t)(i + 1) * FEAT_OUT + (o0 + so);
            #pragma unroll
            for (int j = 0; j < KPERT; ++j) {
                const int k = skb + j * 16;
                vr[j] = (ok_o && k < HID) ? col[(size_t)k * W2COLS] : 0.f;
            }
            xv  = xT[(size_t)(i + 1) * BATCH + tid];
            b2n = ok_oL ? fwb2[(size_t)(i + 1) * FEAT_OUT + o0 + oL] : 0.f;
        }

        // ---- compute on buffer `cur` --------------------------------------
        const uint4* bbuf = (const uint4*)Bs[cur];
        #pragma unroll
        for (int mt = 0; mt < 2; ++mt) {
            floatx8 g = {0.f, 0.f, 0.f, 0.f, 0.f, 0.f, 0.f, 0.f};
            #pragma unroll
            for (int kk = 0; kk < KSTEPS; ++kk) {
                ABFrag bfrag;
                const uint4* bp = bbuf + (oL * 28 + kk * 4 + Kh);
                bfrag.q[0] = bp[0];                        // K = 8*Kh + 0..7
                bfrag.q[1] = bp[2];                        // K = 16 + 8*Kh + 0..7
                g = __builtin_amdgcn_wmma_f32_16x16x32_bf16(
                        false, afrag[mt][kk].v, false, bfrag.v,
                        (short)0, g, false, false);
            }
            // epilogue: acc[b,o] += x[b,i] * relu(g + b2)   (b2cur is already
            // resident — fetched one iteration ago, no wait on the prefetch)
            const int bbase = (wave + mt * 8) * 16 + 8 * Kh;
            #pragma unroll
            for (int j = 0; j < 8; ++j) {
                float gv = g[j] + b2cur;
                gv = gv > 0.f ? gv : 0.f;
                acc[mt][j] = fmaf(xcolS[cur][bbase + j], gv, acc[mt][j]);
            }
        }

        // ---- convert + store i+1 into buffer `cur^1` ----------------------
        if (havenext) {
            unsigned short* bd = Bs[cur ^ 1];
            #pragma unroll
            for (int j = 0; j < KPERT; ++j)
                bd[so * HIDP + (skb + j * 16)] = f2bf(vr[j]);
            xcolS[cur ^ 1][tid] = xv;
        }
        b2cur = b2n;
    }

    // ---- accumulate partials across i-chunks --------------------------------
    if (ok_oL) {
        #pragma unroll
        for (int mt = 0; mt < 2; ++mt) {
            const int bbase = (wave + mt * 8) * 16 + 8 * Kh;
            #pragma unroll
            for (int j = 0; j < 8; ++j)
                atomicAdd(accg + (size_t)(bbase + j) * FEAT_OUT + (o0 + oL),
                          acc[mt][j]);
        }
    }
}

// ---------------------------------------------------------------------------
// Finalize: out = PReLU( BN(acc + fc_b) )
// ---------------------------------------------------------------------------
__global__ void finalize_kernel(const float* __restrict__ accg,
                                const float* __restrict__ fbv,
                                const float* __restrict__ gamma,
                                const float* __restrict__ beta,
                                const float* __restrict__ mean,
                                const float* __restrict__ var,
                                const float* __restrict__ alpha,
                                float* __restrict__ out) {
    int idx = blockIdx.x * blockDim.x + threadIdx.x;
    if (idx >= BATCH * FEAT_OUT) return;
    int o = idx % FEAT_OUT;
    float v = accg[idx] + fbv[idx];
    v = (v - mean[o]) * rsqrtf(var[o] + 1e-5f) * gamma[o] + beta[o];
    float a = alpha[0];
    out[idx] = v >= 0.f ? v : a * v;
}

// ---------------------------------------------------------------------------
extern "C" void kernel_launch(void* const* d_in, const int* in_sizes, int n_in,
                              void* d_out, int out_size, void* d_ws, size_t ws_size,
                              hipStream_t stream) {
    const float* input_emb = (const float*)d_in[0];
    const float* gen_emb   = (const float*)d_in[1];
    const float* cw_W1 = (const float*)d_in[2];
    const float* cw_b1 = (const float*)d_in[3];
    const float* cw_W2 = (const float*)d_in[4];
    const float* cw_b2 = (const float*)d_in[5];
    const float* cb_W1 = (const float*)d_in[6];
    const float* cb_b1 = (const float*)d_in[7];
    const float* cb_W2 = (const float*)d_in[8];
    const float* cb_b2 = (const float*)d_in[9];
    const float* fw_W1 = (const float*)d_in[10];
    const float* fw_b1 = (const float*)d_in[11];
    const float* fw_W2 = (const float*)d_in[12];
    const float* fw_b2 = (const float*)d_in[13];
    const float* fb_W1 = (const float*)d_in[14];
    const float* fb_b1 = (const float*)d_in[15];
    const float* fb_W2 = (const float*)d_in[16];
    const float* fb_b2 = (const float*)d_in[17];
    const float* bn_gamma = (const float*)d_in[18];
    const float* bn_beta  = (const float*)d_in[19];
    const float* bn_mean  = (const float*)d_in[20];
    const float* bn_var   = (const float*)d_in[21];
    const float* prelu_a  = (const float*)d_in[22];
    float* out = (float*)d_out;

    // workspace carve-up (all offsets 256B aligned)
    char* ws = (char*)d_ws;
    float*          ws_convw = (float*)(ws + 0);                 // 256*288 f32
    float*          ws_convb = (float*)(ws + 294912);            // 256*32  f32
    float*          ws_fbv   = (float*)(ws + 327680);            // 256*200 f32
    unsigned short* ws_H     = (unsigned short*)(ws + 532480);   // 256*224 bf16
    float*          ws_xT    = (float*)(ws + 647168);            // 4608*256 f32
    float*          ws_acc   = (float*)(ws + 5365760);           // 256*200 f32

    gen_small_kernel<<<BATCH, 256, 0, stream>>>(gen_emb, cw_W1, cw_b1, cw_W2, cw_b2,
                                                ws_convw, NC * 9);
    gen_small_kernel<<<BATCH, 256, 0, stream>>>(gen_emb, cb_W1, cb_b1, cb_W2, cb_b2,
                                                ws_convb, NC);
    gen_small_kernel<<<BATCH, 256, 0, stream>>>(gen_emb, fb_W1, fb_b1, fb_W2, fb_b2,
                                                ws_fbv, FEAT_OUT);
    fw_hidden_kernel<<<BATCH, 256, 0, stream>>>(gen_emb, fw_W1, fw_b1, ws_H);
    conv_kernel<<<BATCH, 256, 0, stream>>>(input_emb, ws_convw, ws_convb, ws_xT);
    zero_kernel<<<(BATCH * FEAT_OUT + 255) / 256, 256, 0, stream>>>(ws_acc,
                                                                    BATCH * FEAT_OUT);
    fused_einsum_kernel<<<dim3(OTILES, ICHUNKS), 256, 0, stream>>>(
        ws_H, fw_W2, fw_b2, ws_xT, ws_acc);
    finalize_kernel<<<(BATCH * FEAT_OUT + 255) / 256, 256, 0, stream>>>(
        ws_acc, ws_fbv, bn_gamma, bn_beta, bn_mean, bn_var, prelu_a, out);
}
